// GraphSAGE_28802050687442
// MI455X (gfx1250) — matrix operand order repaired and checked by reading, not compiled, not run on hardware
//
#include <hip/hip_runtime.h>
#include <cstddef>

typedef __attribute__((ext_vector_type(16))) __bf16 v16bf;
typedef __attribute__((ext_vector_type(8)))  __bf16 v8bf;
typedef __attribute__((ext_vector_type(8)))  float  v8f;

#define IN_F 128
#define CAT_K 256

// ---------------- zero ----------------
__global__ void k_zero(float* __restrict__ p, size_t n) {
    size_t t = (size_t)blockIdx.x * blockDim.x + threadIdx.x;
    if (t < n) p[t] = 0.0f;
}

// ---------------- weight concat + bf16 pack: catW[o][k] = k<128 ? Wl[o][k] : Wr[o][k-128]
__global__ void k_wpack(const float* __restrict__ Wl, const float* __restrict__ Wr,
                        __bf16* __restrict__ catW, int outf) {
    int t = blockIdx.x * blockDim.x + threadIdx.x;
    if (t >= outf * CAT_K) return;
    int k = t & (CAT_K - 1);
    int o = t >> 8;
    float v = (k < IN_F) ? Wl[o * IN_F + k] : Wr[o * IN_F + (k - IN_F)];
    catW[t] = (__bf16)v;
}

// ---------------- edge scatter, fp32 source (layer 1), one wave per edge ----------------
__global__ void k_scatter_f32(const float* __restrict__ feat, const int* __restrict__ ei,
                              float* __restrict__ sum, float* __restrict__ cnt, int nE) {
    int e    = blockIdx.x * 8 + (threadIdx.x >> 5);
    int lane = threadIdx.x & 31;
    if (e >= nE) return;
    int s = ei[e];
    int d = ei[nE + e];
    const float* fs = feat + (size_t)s * IN_F;
    float*       fd = sum  + (size_t)d * IN_F;
#pragma unroll
    for (int c = 0; c < 4; ++c) {
        int f = lane + 32 * c;
        atomicAdd(&fd[f], fs[f]);
    }
    if (lane == 0) atomicAdd(&cnt[d], 1.0f);
}

// ---------------- edge scatter, bf16 source with row stride CAT_K (layer 2) ----------------
__global__ void k_scatter_bf16(const __bf16* __restrict__ feat, const int* __restrict__ ei,
                               float* __restrict__ sum, int nE) {
    int e    = blockIdx.x * 8 + (threadIdx.x >> 5);
    int lane = threadIdx.x & 31;
    if (e >= nE) return;
    int s = ei[e];
    int d = ei[nE + e];
    const __bf16* fs = feat + (size_t)s * CAT_K;   // pre-offset to second half by caller
    float*        fd = sum  + (size_t)d * IN_F;
#pragma unroll
    for (int c = 0; c < 4; ++c) {
        int f = lane + 32 * c;
        atomicAdd(&fd[f], (float)fs[f]);
    }
}

// ---------------- mean + pack to bf16; optionally also pack x into second half ----------------
__global__ void k_meanpack(const float* __restrict__ sum, const float* __restrict__ cnt,
                           const float* __restrict__ x, __bf16* __restrict__ catA, int nNodes) {
    size_t t = (size_t)blockIdx.x * blockDim.x + threadIdx.x;
    if (t >= (size_t)nNodes * IN_F) return;
    size_t i = t >> 7;
    int    f = (int)(t & (IN_F - 1));
    float  c = cnt[i];
    float  m = sum[t] / fmaxf(c, 1.0f);
    catA[i * CAT_K + f] = (__bf16)m;
    if (x) catA[i * CAT_K + IN_F + f] = (__bf16)x[t];
}

// ---------------- fused concat-GEMM + bias + row L2 norm (+ReLU) via WMMA bf16 ----------------
// Block: (32, outf/16). Each block owns 16 rows; wave w owns cols [16w, 16w+16).
// K = 256 in 8 steps of v_wmma_f32_16x16x32_bf16.
__global__ void k_sage_gemm(const __bf16* __restrict__ catA, const __bf16* __restrict__ catW,
                            const float* __restrict__ bias,
                            float* __restrict__ outF, __bf16* __restrict__ outB,
                            int outf, int relu, int nNodes) {
    __shared__ float sumsq[16];
    const int lane = threadIdx.x;            // 0..31
    const int wave = threadIdx.y;
    const int tid  = wave * 32 + lane;
    const int blk  = blockIdx.x;

    if (tid < 16) sumsq[tid] = 0.0f;
    __syncthreads();

    const int rowm   = lane & 15;
    size_t    row    = (size_t)blk * 16 + rowm;
    if (row >= (size_t)nNodes) row = nNodes - 1;          // clamp (keep EXEC full for WMMA)
    const int kbaseA = (lane < 16) ? 0 : 16;              // A: lanes 0-15 -> K 0..15, 16-31 -> K 16..31
    const int col    = wave * 16 + rowm;                  // B/N column for this lane
    const int kbaseB = (lane < 16) ? 0 : 16;              // B: lanes 0-15 -> K 0..15, 16-31 -> K 16..31

    const __bf16* pa = catA + row * CAT_K;
    const __bf16* pw = catW + (size_t)col * CAT_K;

    v8f acc = {};
#pragma unroll
    for (int kb = 0; kb < 8; ++kb) {
        const int k0 = kb * 32;
        v8bf a0 = *(const v8bf*)(pa + k0 + kbaseA);        // K = kbase .. kbase+7
        v8bf a1 = *(const v8bf*)(pa + k0 + kbaseA + 8);    // K = kbase+8 .. kbase+15
        v16bf A;
#pragma unroll
        for (int i = 0; i < 8; ++i) { A[i] = a0[i]; A[i + 8] = a1[i]; }
        v16bf B = *(const v16bf*)(pw + k0 + kbaseB);       // 16 contiguous K for column `col`
        acc = __builtin_amdgcn_wmma_f32_16x16x32_bf16(
                  false, A, false, B, (short)0, acc, false, false);
    }
    __syncthreads();   // all catA reads complete before any aliased outB writes

    const float b     = bias[col];
    const int   rbase = (lane < 16) ? 0 : 8;               // C layout: VGPR j -> row j / j+8
    float v[8];
#pragma unroll
    for (int j = 0; j < 8; ++j) {
        v[j] = acc[j] + b;
        atomicAdd(&sumsq[rbase + j], v[j] * v[j]);         // ds_add_f32 row reduction
    }
    __syncthreads();

#pragma unroll
    for (int j = 0; j < 8; ++j) {
        const int rm = rbase + j;
        const float s     = sumsq[rm];
        const float scale = 1.0f / fmaxf(sqrtf(s), 1e-12f);
        float val = v[j] * scale;
        if (relu) val = fmaxf(val, 0.0f);
        const size_t R = (size_t)blk * 16 + rm;
        if (R < (size_t)nNodes) {
            if (outF) outF[R * (size_t)outf + col] = val;
            if (outB) outB[R * CAT_K + col] = (__bf16)val;  // outB pre-offset by 128
        }
    }
}

extern "C" void kernel_launch(void* const* d_in, const int* in_sizes, int n_in,
                              void* d_out, int out_size, void* d_ws, size_t ws_size,
                              hipStream_t stream) {
    const float* x   = (const float*)d_in[0];
    const int*   ei  = (const int*)d_in[1];
    const float* W1l = (const float*)d_in[2];
    const float* b1l = (const float*)d_in[3];
    const float* W1r = (const float*)d_in[4];
    const float* W2l = (const float*)d_in[5];
    const float* b2l = (const float*)d_in[6];
    const float* W2r = (const float*)d_in[7];
    float* out = (float*)d_out;

    const int nNodes = in_sizes[0] / IN_F;     // 100000
    const int nE     = in_sizes[1] / 2;        // 1600000

    // workspace layout (all 16B-aligned for these sizes)
    float*  sum    = (float*)d_ws;                              // [N,128] f32
    float*  cnt    = sum + (size_t)nNodes * IN_F;               // [N]     f32
    __bf16* catA   = (__bf16*)(cnt + nNodes);                   // [N,256] bf16: [mean | h]
    __bf16* catW1  = catA + (size_t)nNodes * CAT_K;             // [128,256] bf16
    __bf16* catW2  = catW1 + (size_t)IN_F * CAT_K;              // [64,256]  bf16

    const int TB = 256;
    const size_t nSumCnt = (size_t)nNodes * IN_F + nNodes;
    const size_t nFeat   = (size_t)nNodes * IN_F;
    const int edgeBlocks = (nE + 7) / 8;
    const int rowBlocks  = (nNodes + 15) / 16;

    // ---- weights -> bf16 concat (independent of data path)
    k_wpack<<<(128 * CAT_K + TB - 1) / TB, TB, 0, stream>>>(W1l, W1r, catW1, 128);
    k_wpack<<<(64  * CAT_K + TB - 1) / TB, TB, 0, stream>>>(W2l, W2r, catW2, 64);

    // ---- layer 1
    k_zero<<<(int)((nSumCnt + TB - 1) / TB), TB, 0, stream>>>(sum, nSumCnt);
    k_scatter_f32<<<edgeBlocks, TB, 0, stream>>>(x, ei, sum, cnt, nE);
    k_meanpack<<<(int)((nFeat + TB - 1) / TB), TB, 0, stream>>>(sum, cnt, x, catA, nNodes);
    {
        dim3 blk(32, 8);   // 128 output cols
        k_sage_gemm<<<rowBlocks, blk, 0, stream>>>(catA, catW1, b1l,
                                                   nullptr, catA + IN_F, 128, 1, nNodes);
    }

    // ---- layer 2 (counts unchanged; h lives in catA[:,128:256] as bf16)
    k_zero<<<(int)((nFeat + TB - 1) / TB), TB, 0, stream>>>(sum, nFeat);
    k_scatter_bf16<<<edgeBlocks, TB, 0, stream>>>(catA + IN_F, ei, sum, nE);
    k_meanpack<<<(int)((nFeat + TB - 1) / TB), TB, 0, stream>>>(sum, cnt, nullptr, catA, nNodes);
    {
        dim3 blk(32, 4);   // 64 output cols
        k_sage_gemm<<<rowBlocks, blk, 0, stream>>>(catA, catW2, b2l,
                                                   out, nullptr, 64, 0, nNodes);
    }
}
